// LocalSelfAttention_3487513444450
// MI455X (gfx1250) — matrix operand order repaired and checked by reading, not compiled, not run on hardware
//
#include <hip/hip_runtime.h>

// ---------------------------------------------------------------------------
// Problem constants (from reference): b=4, S=2048, D=1024, H=16, DK=DV=64, K=7
// ---------------------------------------------------------------------------
#define BATCH   4
#define SEQ     2048
#define DMODEL  1024
#define NHEAD   16
#define HDIM    64
#define MROWS   (BATCH * SEQ)        // 8192

// GEMM tiling: 128x128 block tile, K-step 64 (2 WMMA k-steps per barrier pair)
#define TM 128
#define TN 128
#define TK 64
#define KP 72                        // padded LDS row (halfs): 144B, 16B-aligned, bank-spread

typedef __attribute__((ext_vector_type(8)))  float    v8f;
typedef __attribute__((ext_vector_type(8)))  _Float16 v8h;
typedef __attribute__((ext_vector_type(16))) _Float16 v16h;

__device__ __forceinline__ v8h cvt8(const float4& a, const float4& b) {
    v8h r;
    r[0] = (_Float16)a.x; r[1] = (_Float16)a.y; r[2] = (_Float16)a.z; r[3] = (_Float16)a.w;
    r[4] = (_Float16)b.x; r[5] = (_Float16)b.y; r[6] = (_Float16)b.z; r[7] = (_Float16)b.w;
    return r;
}

// ---------------------------------------------------------------------------
// One-time weight prep: WT16[n][k] = (fp16) W[k][n].  Tiled 32x32 transpose.
// Amortized over 64 block-rows per GEMM; WT16 (2MB) stays L2-resident.
// ---------------------------------------------------------------------------
__global__ __launch_bounds__(256) void transpose_cvt_kernel(
    const float* __restrict__ W, _Float16* __restrict__ WT, int rows, int cols)
{
    __shared__ float T[32][33];
    const int tx = threadIdx.x & 31;
    const int ty = threadIdx.x >> 5;          // 0..7
    const int kBase = blockIdx.x * 32;        // source row (K)
    const int nBase = blockIdx.y * 32;        // source col (N)
#pragma unroll
    for (int j = 0; j < 32; j += 8)
        T[ty + j][tx] = W[(size_t)(kBase + ty + j) * cols + nBase + tx];
    __syncthreads();
#pragma unroll
    for (int j = 0; j < 32; j += 8)
        WT[(size_t)(nBase + ty + j) * rows + kBase + tx] = (_Float16)T[tx][ty + j];
}

// ---------------------------------------------------------------------------
// C[M,N] = A[M,K] @ W[K,N] + bias[N] (+ residual[M,N])
// A: f32 row-major (converted to fp16 while staging into LDS).
// WT: fp16, PRE-TRANSPOSED [N][K] -> pure b128 copy into LDS.
// f32 accumulation via v_wmma_f32_16x16x32_f16.
// Block: 256 threads = 8 waves; wave owns 32x64 of the 128x128 tile.
// ---------------------------------------------------------------------------
__global__ __launch_bounds__(256) void wmma_gemm_bias(
    const float* __restrict__ A, const _Float16* __restrict__ WT,
    const float* __restrict__ bias, const float* __restrict__ residual,
    float* __restrict__ C, int M, int N, int K)
{
    __shared__ _Float16 As[TM][KP];    // A tile, row-major (m, k)
    __shared__ _Float16 BsT[TN][KP];   // W tile, transposed (n, k)

    const int tid  = threadIdx.x;
    const int lane = tid & 31;
    const int wave = tid >> 5;
    const int half = lane >> 4;        // 0: lanes 0-15, 1: lanes 16-31
    const int l16  = lane & 15;
    const int wm   = (wave & 3) * 32;  // wave row offset in block tile
    const int wn   = (wave >> 2) * 64; // wave col offset in block tile

    const int mBase = blockIdx.y * TM;
    const int nBase = blockIdx.x * TN;

    v8f acc[2][4];
#pragma unroll
    for (int mt = 0; mt < 2; ++mt)
#pragma unroll
        for (int nt = 0; nt < 4; ++nt)
            acc[mt][nt] = (v8f){0.f, 0.f, 0.f, 0.f, 0.f, 0.f, 0.f, 0.f};

    // staging indices: 2 threads per row, each covers 32 halfs (64B)
    const int srow = tid >> 1;             // 0..127
    const int scs  = (tid & 1) << 5;       // 0 or 32

    for (int k0 = 0; k0 < K; k0 += TK) {
        // ---- stage A tile (f32 -> fp16), row-major ----
        {
            const float* ap = A + (size_t)(mBase + srow) * K + k0 + scs;
#pragma unroll
            for (int j = 0; j < 4; ++j) {
                float4 f0 = *(const float4*)(ap + j * 8);
                float4 f1 = *(const float4*)(ap + j * 8 + 4);
                *(v8h*)&As[srow][scs + j * 8] = cvt8(f0, f1);
            }
        }
        // ---- stage W tile: straight fp16 b128 copy (already transposed) ----
        {
            const _Float16* wp = WT + (size_t)(nBase + srow) * K + k0 + scs;
#pragma unroll
            for (int j = 0; j < 4; ++j)
                *(v8h*)&BsT[srow][scs + j * 8] = *(const v8h*)(wp + j * 8);
        }
        if (k0 + TK < K) {  // global_prefetch_b8 of next K tile
            __builtin_prefetch(A  + (size_t)(mBase + srow) * K + k0 + TK + scs, 0, 1);
            __builtin_prefetch(WT + (size_t)(nBase + srow) * K + k0 + TK + scs, 0, 1);
        }
        __syncthreads();

#pragma unroll
        for (int ks = 0; ks < 2; ++ks) {
            const int kb = ks << 5;        // 0 or 32

            // A fragment: 16x32 fp16 layout (elem e -> K = (e>>3)*16 + half*8 + (e&7))
            v16h af[2];
#pragma unroll
            for (int mt = 0; mt < 2; ++mt) {
                const _Float16* ar = &As[wm + mt * 16 + l16][kb];
                v8h lo = *(const v8h*)(ar + (half << 3));
                v8h hi = *(const v8h*)(ar + 16 + (half << 3));
                v16h a;
#pragma unroll
                for (int e = 0; e < 8; ++e) { a[e] = lo[e]; a[e + 8] = hi[e]; }
                af[mt] = a;
            }
            // B fragment: 32x16 fp16 layout (elem e -> K = (e>>3)*16 + half*8 + (e&7), N = l16)
            v16h bf[4];
#pragma unroll
            for (int nt = 0; nt < 4; ++nt) {
                const _Float16* br = &BsT[wn + nt * 16 + l16][kb + (half << 3)];
                v8h lo = *(const v8h*)(br);
                v8h hi = *(const v8h*)(br + 16);
                v16h b;
#pragma unroll
                for (int e = 0; e < 8; ++e) { b[e] = lo[e]; b[e + 8] = hi[e]; }
                bf[nt] = b;
            }

#pragma unroll
            for (int mt = 0; mt < 2; ++mt)
#pragma unroll
                for (int nt = 0; nt < 4; ++nt)
                    acc[mt][nt] = __builtin_amdgcn_wmma_f32_16x16x32_f16(
                        false, af[mt], false, bf[nt], (short)0, acc[mt][nt], false, false);
        }
        __syncthreads();
    }

    // ---- epilogue: C/D layout reg j -> M = half*8 + j, N = l16 ----
#pragma unroll
    for (int nt = 0; nt < 4; ++nt) {
        const int col = nBase + wn + nt * 16 + l16;
        const float bv = bias[col];
#pragma unroll
        for (int mt = 0; mt < 2; ++mt) {
#pragma unroll
            for (int j = 0; j < 8; ++j) {
                const int row = mBase + wm + mt * 16 + half * 8 + j;
                float v = acc[mt][nt][j] + bv;
                if (residual) v += residual[(size_t)row * N + col];
                C[(size_t)row * N + col] = v;
            }
        }
    }
}

// ---------------------------------------------------------------------------
// Windowed attention: one wave per (b,s,h). Lane covers 2 of the 64 dims.
// k/v for window offset i come from kk/vv at position s+i-3; out-of-range
// positions project the zero padding -> bias vector only.
// o may alias q (each lane reads q before writing o at the same address).
// ---------------------------------------------------------------------------
__global__ __launch_bounds__(256) void attn_kernel(
    const float* __restrict__ q, const float* __restrict__ kk,
    const float* __restrict__ vv, const float* __restrict__ b_ks,
    const float* __restrict__ b_vs, float* __restrict__ o)
{
    const int lane = threadIdx.x & 31;
    const int wave = threadIdx.x >> 5;
    const int idx  = blockIdx.x * 8 + wave;      // (b*S + s)*H + h
    const int h    = idx & (NHEAD - 1);
    const int bs   = idx >> 4;                   // b*S + s
    const int s    = bs & (SEQ - 1);
    const int d0   = lane << 1;

    const size_t qoff = (size_t)idx * HDIM + d0;
    const float2 q2 = *(const float2*)&q[qoff];

    float  sc[7];
    float2 vreg[7];
#pragma unroll
    for (int i = 0; i < 7; ++i) {
        const int pos = s + i - 3;
        float2 k2, v2;
        if (pos >= 0 && pos < SEQ) {
            const size_t off = ((size_t)(bs - s + pos) * NHEAD + h) * HDIM + d0;
            k2 = *(const float2*)&kk[off];
            v2 = *(const float2*)&vv[off];
        } else {
            k2 = *(const float2*)&b_ks[h * HDIM + d0];
            v2 = *(const float2*)&b_vs[h * HDIM + d0];
        }
        float p = q2.x * k2.x + q2.y * k2.y;
#pragma unroll
        for (int m = 16; m >= 1; m >>= 1) p += __shfl_xor(p, m, 32);
        sc[i]   = p * 0.125f;                    // / sqrt(DK) = / 8
        vreg[i] = v2;
    }

    float mx = sc[0];
#pragma unroll
    for (int i = 1; i < 7; ++i) mx = fmaxf(mx, sc[i]);
    float sum = 0.f;
#pragma unroll
    for (int i = 0; i < 7; ++i) { sc[i] = __expf(sc[i] - mx); sum += sc[i]; }
    const float inv = 1.0f / sum;

    float2 out; out.x = 0.f; out.y = 0.f;
#pragma unroll
    for (int i = 0; i < 7; ++i) {
        out.x += sc[i] * vreg[i].x;
        out.y += sc[i] * vreg[i].y;
    }
    out.x *= inv; out.y *= inv;
    *(float2*)&o[qoff] = out;
}

// ---------------------------------------------------------------------------
// In-place LayerNorm over D=1024 per row. One block (256 thr) per row.
// ---------------------------------------------------------------------------
__global__ __launch_bounds__(256) void ln_kernel(
    float* __restrict__ y, const float* __restrict__ g, const float* __restrict__ b)
{
    __shared__ float red[16];
    const int tid  = threadIdx.x;
    const int lane = tid & 31;
    const int wave = tid >> 5;
    float* yr = y + (size_t)blockIdx.x * DMODEL;

    float4 v = *(const float4*)&yr[tid * 4];
    float s  = v.x + v.y + v.z + v.w;
    float ss = v.x * v.x + v.y * v.y + v.z * v.z + v.w * v.w;
#pragma unroll
    for (int m = 16; m >= 1; m >>= 1) {
        s  += __shfl_xor(s,  m, 32);
        ss += __shfl_xor(ss, m, 32);
    }
    if (lane == 0) { red[wave] = s; red[8 + wave] = ss; }
    __syncthreads();
    s = 0.f; ss = 0.f;
#pragma unroll
    for (int w = 0; w < 8; ++w) { s += red[w]; ss += red[8 + w]; }

    const float mu  = s * (1.0f / DMODEL);
    const float var = ss * (1.0f / DMODEL) - mu * mu;
    const float r   = rsqrtf(var + 1e-5f);

    const float4 gv = *(const float4*)&g[tid * 4];
    const float4 bv = *(const float4*)&b[tid * 4];
    float4 o;
    o.x = (v.x - mu) * r * gv.x + bv.x;
    o.y = (v.y - mu) * r * gv.y + bv.y;
    o.z = (v.z - mu) * r * gv.z + bv.z;
    o.w = (v.w - mu) * r * gv.w + bv.w;
    *(float4*)&yr[tid * 4] = o;
}

// ---------------------------------------------------------------------------
extern "C" void kernel_launch(void* const* d_in, const int* in_sizes, int n_in,
                              void* d_out, int out_size, void* d_ws, size_t ws_size,
                              hipStream_t stream)
{
    (void)in_sizes; (void)n_in; (void)out_size; (void)ws_size;

    const float* x    = (const float*)d_in[0];
    const float* xs   = (const float*)d_in[1];
    // d_in[2] = xs_ : unused by the reference
    const float* w_qs = (const float*)d_in[3];
    const float* b_qs = (const float*)d_in[4];
    const float* w_ks = (const float*)d_in[5];
    const float* b_ks = (const float*)d_in[6];
    const float* w_vs = (const float*)d_in[7];
    const float* b_vs = (const float*)d_in[8];
    const float* w_fc = (const float*)d_in[9];
    const float* b_fc = (const float*)d_in[10];
    const float* ln_g = (const float*)d_in[11];
    const float* ln_b = (const float*)d_in[12];

    const size_t NW = (size_t)DMODEL * DMODEL;      // 1M elements per weight

    float* q  = (float*)d_ws;                       // 8192*1024 f32 (32 MB)
    float* kk = q  + (size_t)MROWS * DMODEL;        // 32 MB
    float* vv = kk + (size_t)MROWS * DMODEL;        // 32 MB
    _Float16* wqT  = (_Float16*)(vv + (size_t)MROWS * DMODEL);  // 4 x 2 MB fp16
    _Float16* wkT  = wqT + NW;
    _Float16* wvT  = wkT + NW;
    _Float16* wfcT = wvT + NW;
    float* y  = (float*)d_out;

    // one-time fp16 transposed weights (L2-resident, reused by 64 block-rows)
    const dim3 gt(DMODEL / 32, DMODEL / 32);        // (32, 32)
    transpose_cvt_kernel<<<gt, 256, 0, stream>>>(w_qs, wqT,  DMODEL, DMODEL);
    transpose_cvt_kernel<<<gt, 256, 0, stream>>>(w_ks, wkT,  DMODEL, DMODEL);
    transpose_cvt_kernel<<<gt, 256, 0, stream>>>(w_vs, wvT,  DMODEL, DMODEL);
    transpose_cvt_kernel<<<gt, 256, 0, stream>>>(w_fc, wfcT, DMODEL, DMODEL);

    const dim3 gg(DMODEL / TN, MROWS / TM);         // (8, 64)

    // Q/K/V projections: project xs ONCE (window gather happens in attention)
    wmma_gemm_bias<<<gg, 256, 0, stream>>>(x,  wqT, b_qs, nullptr, q,  MROWS, DMODEL, DMODEL);
    wmma_gemm_bias<<<gg, 256, 0, stream>>>(xs, wkT, b_ks, nullptr, kk, MROWS, DMODEL, DMODEL);
    wmma_gemm_bias<<<gg, 256, 0, stream>>>(xs, wvT, b_vs, nullptr, vv, MROWS, DMODEL, DMODEL);

    // windowed attention; writes o in-place over q (safe: per-lane read-then-write)
    attn_kernel<<<(BATCH * SEQ * NHEAD) / 8, 256, 0, stream>>>(q, kk, vv, b_ks, b_vs, q);

    // fc GEMM fused with bias + residual(x) -> d_out
    wmma_gemm_bias<<<gg, 256, 0, stream>>>(q, wfcT, b_fc, x, y, MROWS, DMODEL, DMODEL);

    // in-place LayerNorm
    ln_kernel<<<MROWS, 256, 0, stream>>>(y, ln_g, ln_b);
}
